// PTBModel_44212393345118
// MI455X (gfx1250) — matrix operand order repaired and checked by reading, not compiled, not run on hardware
//
#include <hip/hip_runtime.h>
#include <hip/hip_bf16.h>

// ---------------------------------------------------------------------------
// PTB LSTM LM on MI455X (gfx1250, wave32, WMMA).
// bf16 WMMA (f32 accumulate) everywhere; weights transposed to [N,K] bf16 so
// B operands read contiguous K. LSTM: fused 4-gate waves, per-step kernels.
// Projection: workgroup-shared B tiles staged into LDS with
// global_load_async_to_lds_b128 (double-buffered, ASYNCcnt + barrier), waves
// read B fragments from LDS (ds_load_b128) and run 8 WMMAs per K-step.
// ---------------------------------------------------------------------------

typedef __bf16 bf16_t;
typedef bf16_t v8bf  __attribute__((ext_vector_type(8)));
typedef bf16_t v16bf __attribute__((ext_vector_type(16)));
typedef float  v8f   __attribute__((ext_vector_type(8)));

#define VOCAB 32000
#define EMB   1024
#define HID   1024
#define SEQ   128
#define BATCH 32

__device__ __forceinline__ float sigmoid_(float x) {
    return 1.0f / (1.0f + __expf(-x));
}
__device__ __forceinline__ float tanh_(float x) {
    float e = __expf(2.0f * x);
    return (e - 1.0f) / (e + 1.0f);
}

// Concatenate two 8-wide bf16 chunks into the 16-wide WMMA A/B operand.
__device__ __forceinline__ v16bf cat16(v8bf lo, v8bf hi) {
    return __builtin_shufflevector(lo, hi, 0,1,2,3,4,5,6,7,8,9,10,11,12,13,14,15);
}

// Generic (flat) shared-memory pointer -> 32-bit LDS byte address.
__device__ __forceinline__ uint32_t lds_addr(const void* p) {
    return (uint32_t)(uintptr_t)(__attribute__((address_space(3))) const char*)p;
}

// Per-lane async copy of 16 bytes global -> LDS (tracked by ASYNCcnt).
__device__ __forceinline__ void async_g2l_b128(uint32_t lds_byte_addr, const void* gaddr) {
    asm volatile("global_load_async_to_lds_b128 %0, %1, off"
                 :: "v"(lds_byte_addr), "v"(gaddr) : "memory");
}
__device__ __forceinline__ void wait_async0() {
    asm volatile("s_wait_asynccnt 0x0" ::: "memory");
}

// ---------------------------------------------------------------------------
// Embedding gather + f32->bf16:  x[t,b,:] = bf16(emb[seq[t,b], :])
// ---------------------------------------------------------------------------
__global__ void k_embed(const int* __restrict__ seq, const float* __restrict__ emb,
                        bf16_t* __restrict__ x) {
    int i   = blockIdx.x * blockDim.x + threadIdx.x;   // over SEQ*BATCH*EMB
    int col = i & (EMB - 1);
    int rb  = i >> 10;                                  // t*BATCH + b
    int tok = seq[rb];
    x[i] = (bf16_t)emb[(size_t)tok * EMB + col];
}

// ---------------------------------------------------------------------------
// Weight convert + transpose:  Wt[n,k] = bf16(W[k,n]),  W is [K,N] row-major.
// ---------------------------------------------------------------------------
__global__ void k_w_to_bf16_t(const float* __restrict__ W, bf16_t* __restrict__ Wt,
                              int K, int N) {
    int i = blockIdx.x * blockDim.x + threadIdx.x;
    int n = i % N;
    int k = i / N;
    if (k < K) Wt[(size_t)n * K + k] = (bf16_t)W[i];
}

// ---------------------------------------------------------------------------
// Zero LSTM states (h double buffers bf16, c f32), BATCH*HID each.
// ---------------------------------------------------------------------------
__global__ void k_zero_state(bf16_t* h0a, bf16_t* h0b, float* c0,
                             bf16_t* h1a, bf16_t* h1b, float* c1) {
    int i = blockIdx.x * blockDim.x + threadIdx.x;     // BATCH*HID = 32768
    h0a[i] = (bf16_t)0.0f; h0b[i] = (bf16_t)0.0f; c0[i] = 0.0f;
    h1a[i] = (bf16_t)0.0f; h1b[i] = (bf16_t)0.0f; c1[i] = 0.0f;
}

// ---------------------------------------------------------------------------
// One LSTM timestep. Grid: 16 blocks x 256 thr = 128 waves.
// Each wave: 16x16 tile of all four gates (i,j,f,o) for its
// (batch-tile, hidden-tile); A fragment shared by 4 WMMAs; cell update fused.
// ---------------------------------------------------------------------------
__global__ void __launch_bounds__(256) k_lstm_step(
    const bf16_t* __restrict__ xseq,   // [SEQ, BATCH, HID] layer input (bf16)
    const bf16_t* __restrict__ Wt,     // [4H, 2H] bf16, row n = output col n
    const float*  __restrict__ bias,   // [4H]
    const bf16_t* __restrict__ h_in,   // [BATCH, HID] h_{t-1}
    bf16_t*       __restrict__ h_out,  // [BATCH, HID] h_t
    float*        __restrict__ c_st,   // [BATCH, HID] cell state (in/out)
    bf16_t*       __restrict__ yout,   // [SEQ, BATCH, HID] layer output
    int t)
{
    const int K = 2 * HID;
    int wave   = (blockIdx.x * blockDim.x + threadIdx.x) >> 5;
    int lane   = threadIdx.x & 31;
    int laneHi = lane >> 4;
    int lanelo = lane & 15;
    int m_tile = wave & 1;         // batch tile (2 x 16 = 32)
    int htile  = wave >> 1;        // hidden tile (64 x 16 = 1024)
    int brow_m = m_tile * 16 + lanelo;   // A-matrix row this lane feeds
    int hid0   = htile * 16;

    const bf16_t* xrow = xseq + ((size_t)t * BATCH + brow_m) * HID;
    const bf16_t* hrow = h_in + (size_t)brow_m * HID;

    const bf16_t* wrow[4];
#pragma unroll
    for (int g = 0; g < 4; ++g)
        wrow[g] = Wt + (size_t)(g * HID + hid0 + lanelo) * K;

    v8f acc[4] = {v8f{}, v8f{}, v8f{}, v8f{}};   // gates i, j, f, o

    for (int k0 = 0; k0 < K; k0 += 32) {
        // A fragment (16x32 bf16): lane holds row brow_m,
        // 16 contiguous K starting at ka = k0 + 16*laneHi
        int ka = k0 + laneHi * 16;
        const bf16_t* asrc = (k0 < HID) ? (xrow + ka) : (hrow + (ka - HID));
        v16bf a = *(const v16bf*)(asrc);

        // B fragment (32x16): lane covers column hid0+lanelo of its gate,
        // 16 contiguous K starting at k0 + 16*laneHi.
        int kb = k0 + laneHi * 16;
#pragma unroll
        for (int g = 0; g < 4; ++g) {
            v16bf b = *(const v16bf*)(wrow[g] + kb);
            acc[g] = __builtin_amdgcn_wmma_f32_16x16x32_bf16(
                false, a, false, b, (short)0, acc[g], false, false);
        }
    }

    // Cell update.  C/D layout: VGPR e -> M = e + 8*laneHi, N = lanelo.
    int hid = hid0 + lanelo;
    float bi = bias[hid];
    float bj = bias[HID + hid];
    float bf = bias[2 * HID + hid];
    float bo = bias[3 * HID + hid];
#pragma unroll
    for (int e = 0; e < 8; ++e) {
        int m  = e + laneHi * 8;
        int bb = m_tile * 16 + m;                 // batch index
        size_t off = (size_t)bb * HID + hid;
        float zi = acc[0][e] + bi;
        float zj = acc[1][e] + bj;
        float zf = acc[2][e] + bf;
        float zo = acc[3][e] + bo;
        float c_old = c_st[off];
        float c2 = c_old * sigmoid_(zf + 1.0f) + sigmoid_(zi) * tanh_(zj);
        float h2 = tanh_(c2) * sigmoid_(zo);
        c_st[off]  = c2;
        bf16_t hb  = (bf16_t)h2;
        h_out[off] = hb;
        yout[((size_t)t * BATCH + bb) * HID + hid] = hb;
    }
}

// ---------------------------------------------------------------------------
// Projection: logits[4096,32000] = Y[4096,1024] @ Wd[1024,32000] + bd.
// Wdt is [32000,1024] bf16.  Workgroup (8 waves) computes 256(M) x 64(N);
// the 64x32 bf16 B tile is shared by all waves -> staged into LDS with
// global_load_async_to_lds_b128, double buffered.  Each wave: 32x64 tile
// (2 A fragments, 4 LDS B fragments, 8 WMMAs per K-step).
// Grid: 16 M-blocks x 500 N-blocks = 8000 WGs.
// ---------------------------------------------------------------------------
__global__ void __launch_bounds__(256) k_proj(
    const bf16_t* __restrict__ Y,     // [SEQ*BATCH, HID] bf16
    const bf16_t* __restrict__ Wdt,   // [VOCAB, HID] bf16
    const float*  __restrict__ bd,    // [VOCAB]
    float*        __restrict__ out)   // [SEQ*BATCH, VOCAB]
{
    __shared__ bf16_t Bs[2][64][32];  // [buf][n_local][k_local] : 8 KB

    int tid    = threadIdx.x;
    int wave   = tid >> 5;
    int lane   = tid & 31;
    int laneHi = lane >> 4;
    int lanelo = lane & 15;
    int wg_m   = blockIdx.x & 15;     // 16 M blocks of 256 rows
    int n_blk  = blockIdx.x >> 4;     // 500 N blocks of 64 cols
    int n0     = n_blk * 64;
    int row_base = wg_m * 256 + wave * 32;

    // Cooperative B staging: thread -> (row sr, 8 cols at sc): one b128 each.
    int sr = tid >> 2;
    int sc = (tid & 3) * 8;
    const bf16_t* gsrc = Wdt + (size_t)(n0 + sr) * HID + sc;
    uint32_t ldst[2] = { lds_addr(&Bs[0][sr][sc]), lds_addr(&Bs[1][sr][sc]) };

    const bf16_t* arow0 = Y + (size_t)(row_base + lanelo) * HID;
    const bf16_t* arow1 = Y + (size_t)(row_base + 16 + lanelo) * HID;

    v8f acc[2][4];
#pragma unroll
    for (int mt = 0; mt < 2; ++mt)
#pragma unroll
        for (int g = 0; g < 4; ++g) acc[mt][g] = v8f{};

    async_g2l_b128(ldst[0], gsrc);                      // prologue: k0 = 0
    for (int k0 = 0; k0 < HID; k0 += 32) {
        int buf = (k0 >> 5) & 1;
        wait_async0();                                  // my slice landed
        __syncthreads();                                // everyone's slice landed
        if (k0 + 32 < HID)
            async_g2l_b128(ldst[buf ^ 1], gsrc + k0 + 32);  // prefetch next

        int ka = k0 + laneHi * 16;
        v16bf a0 = cat16(*(const v8bf*)(arow0 + ka), *(const v8bf*)(arow0 + ka + 8));
        v16bf a1 = cat16(*(const v8bf*)(arow1 + ka), *(const v8bf*)(arow1 + ka + 8));
#pragma unroll
        for (int g = 0; g < 4; ++g) {
            const bf16_t* bp = &Bs[buf][g * 16 + lanelo][laneHi * 16];
            v16bf b = cat16(*(const v8bf*)(bp), *(const v8bf*)(bp + 8));
            acc[0][g] = __builtin_amdgcn_wmma_f32_16x16x32_bf16(
                false, a0, false, b, (short)0, acc[0][g], false, false);
            acc[1][g] = __builtin_amdgcn_wmma_f32_16x16x32_bf16(
                false, a1, false, b, (short)0, acc[1][g], false, false);
        }
        __syncthreads();                                // buf consumed, reusable
    }

#pragma unroll
    for (int g = 0; g < 4; ++g) {
        int col = n0 + g * 16 + lanelo;
        float bc = bd[col];
#pragma unroll
        for (int mt = 0; mt < 2; ++mt) {
#pragma unroll
            for (int e = 0; e < 8; ++e) {
                int r = row_base + mt * 16 + e + laneHi * 8;
                out[(size_t)r * VOCAB + col] = acc[mt][g][e] + bc;
            }
        }
    }
}

// ---------------------------------------------------------------------------
extern "C" void kernel_launch(void* const* d_in, const int* in_sizes, int n_in,
                              void* d_out, int out_size, void* d_ws, size_t ws_size,
                              hipStream_t stream) {
    (void)in_sizes; (void)n_in; (void)out_size; (void)ws_size;

    const int*   seq = (const int*)  d_in[0];
    const float* emb = (const float*)d_in[1];
    const float* W0  = (const float*)d_in[2];
    const float* b0  = (const float*)d_in[3];
    const float* W1  = (const float*)d_in[4];
    const float* b1  = (const float*)d_in[5];
    const float* Wd  = (const float*)d_in[6];
    const float* bd  = (const float*)d_in[7];
    float* out = (float*)d_out;

    // Workspace carve-up (256B aligned).
    char* w = (char*)d_ws;
    auto alloc = [&](size_t bytes) {
        char* p = w;
        w += (bytes + 255) & ~(size_t)255;
        return p;
    };
    const size_t TBH = (size_t)SEQ * BATCH * HID;          // 4,194,304
    bf16_t* xemb = (bf16_t*)alloc(TBH * sizeof(bf16_t));
    bf16_t* y1   = (bf16_t*)alloc(TBH * sizeof(bf16_t));
    bf16_t* y2   = (bf16_t*)alloc(TBH * sizeof(bf16_t));
    bf16_t* Wt0  = (bf16_t*)alloc((size_t)4 * HID * 2 * HID * sizeof(bf16_t));
    bf16_t* Wt1  = (bf16_t*)alloc((size_t)4 * HID * 2 * HID * sizeof(bf16_t));
    bf16_t* Wdt  = (bf16_t*)alloc((size_t)VOCAB * HID * sizeof(bf16_t));
    bf16_t* h0a  = (bf16_t*)alloc((size_t)BATCH * HID * sizeof(bf16_t));
    bf16_t* h0b  = (bf16_t*)alloc((size_t)BATCH * HID * sizeof(bf16_t));
    bf16_t* h1a  = (bf16_t*)alloc((size_t)BATCH * HID * sizeof(bf16_t));
    bf16_t* h1b  = (bf16_t*)alloc((size_t)BATCH * HID * sizeof(bf16_t));
    float*  c0   = (float*) alloc((size_t)BATCH * HID * sizeof(float));
    float*  c1   = (float*) alloc((size_t)BATCH * HID * sizeof(float));

    // 1) Precompute bf16 operands.
    k_embed<<<(SEQ * BATCH * EMB) / 256, 256, 0, stream>>>(seq, emb, xemb);
    k_w_to_bf16_t<<<(2 * HID * 4 * HID) / 256, 256, 0, stream>>>(W0, Wt0, 2 * HID, 4 * HID);
    k_w_to_bf16_t<<<(2 * HID * 4 * HID) / 256, 256, 0, stream>>>(W1, Wt1, 2 * HID, 4 * HID);
    k_w_to_bf16_t<<<(HID * VOCAB) / 256, 256, 0, stream>>>(Wd, Wdt, HID, VOCAB);
    k_zero_state<<<(BATCH * HID) / 256, 256, 0, stream>>>(h0a, h0b, c0, h1a, h1b, c1);

    // 2) LSTM layer 1 (recurrent over T; h double-buffered across steps).
    for (int t = 0; t < SEQ; ++t) {
        const bf16_t* hin  = (t & 1) ? h0b : h0a;
        bf16_t*       hout = (t & 1) ? h0a : h0b;
        k_lstm_step<<<16, 256, 0, stream>>>(xemb, Wt0, b0, hin, hout, c0, y1, t);
    }
    // 3) LSTM layer 2.
    for (int t = 0; t < SEQ; ++t) {
        const bf16_t* hin  = (t & 1) ? h1b : h1a;
        bf16_t*       hout = (t & 1) ? h1a : h1b;
        k_lstm_step<<<16, 256, 0, stream>>>(y1, Wt1, b1, hin, hout, c1, y2, t);
    }

    // 4) Output projection (16 M-blocks x 500 N-blocks).
    k_proj<<<16 * 500, 256, 0, stream>>>(y2, Wdt, bd, out);
}
